// DRGAN_80985903333882
// MI455X (gfx1250) — matrix operand order repaired and checked by reading, not compiled, not run on hardware
//
#include <hip/hip_runtime.h>
#include <hip/hip_bf16.h>

// ---------------------------------------------------------------------------
// GraphSAGE GNN pipeline for MI455X (gfx1250, wave32, WMMA).
//   N=50000 nodes, E=800000 edges, D=128, H=256, C=40, B=3 blocks, G=2 gates.
//
// GEMMs: weights pre-converted once per launch to bf16 transposed [Hout][K];
// each 256-thread block async-copies a K x 64 B-tile into LDS (ASYNCcnt),
// pad-8 rows for conflict-free ds_load_b128. Each wave computes a 16x64
// output strip (4 accumulators) so one A-fragment feeds 4 WMMAs -> 4x less
// A traffic. v_wmma_f32_16x16x32_bf16, f32 accumulate. Scatter phase is
// L2-resident (51MB features vs 192MB L2) f32 atomics.
// ---------------------------------------------------------------------------

#define kN 50000
#define kE 800000
#define kD 128
#define kH 256
#define kC 40
#define kB 3
#define kEPS 1e-5f

typedef __attribute__((ext_vector_type(16))) __bf16 v16bf;
typedef __attribute__((ext_vector_type(8)))  float  v8f;

#ifndef __has_builtin
#define __has_builtin(x) 0
#endif
#if __has_builtin(__builtin_amdgcn_global_load_async_to_lds_b128)
#define HAVE_ASYNC_LDS 1
#else
#define HAVE_ASYNC_LDS 0
#endif

#if HAVE_ASYNC_LDS
// builtin takes v4i pointers: global (AS1) source, LDS (AS3) destination
typedef int vs4i __attribute__((__vector_size__(4 * sizeof(int))));
typedef __attribute__((address_space(1))) vs4i gvs4i;
typedef __attribute__((address_space(3))) vs4i lvs4i;
__device__ inline void async_cp16(const __bf16* g, __bf16* l) {
  __builtin_amdgcn_global_load_async_to_lds_b128(
      (gvs4i*)(const void*)g, (lvs4i*)(void*)l, 0, 0);
}
__device__ inline void wait_async0() {
#if __has_builtin(__builtin_amdgcn_s_wait_asynccnt)
  __builtin_amdgcn_s_wait_asynccnt(0);
#else
  asm volatile("s_wait_asynccnt 0x0" ::: "memory");
#endif
}
#endif

// --- weight prep: W[K,Hout] f32  ->  Wt[Hout,K] bf16 -----------------------
__global__ void transpose_bf16_kernel(const float* __restrict__ W,
                                      __bf16* __restrict__ Wt, int K, int Hout) {
  const int idx = blockIdx.x * blockDim.x + threadIdx.x;
  if (idx >= K * Hout) return;
  const int k = idx / Hout, h = idx % Hout;   // coalesced read of W
  Wt[(size_t)h * K + k] = (__bf16)W[idx];
}

// --- A fragment: 16x32 bf16 from f32 activations ---------------------------
__device__ inline v16bf load_a_frag(const float* __restrict__ A, int lda,
                                    int row, int k0, int lane, float scale) {
  const int half = lane >> 4;
  const float* p = A + (size_t)row * lda + k0 + 16 * half;
  const float4 x0 = ((const float4*)p)[0];
  const float4 x1 = ((const float4*)p)[1];
  const float4 x2 = ((const float4*)(p + 8))[0];
  const float4 x3 = ((const float4*)(p + 8))[1];
  const float xs[16] = {x0.x, x0.y, x0.z, x0.w, x1.x, x1.y, x1.z, x1.w,
                        x2.x, x2.y, x2.z, x2.w, x3.x, x3.y, x3.z, x3.w};
  union { v16bf v; __bf16 e[16]; } u;
#pragma unroll
  for (int j = 0; j < 16; ++j) u.e[j] = (__bf16)(xs[j] * scale);
  return u.v;
}

// --- B fragment from LDS tile (two 16B ds loads) ---------------------------
__device__ inline v16bf lds_b_frag(const __bf16* p) {
  union { v16bf v; float4 f[2]; } u;
  u.f[0] = ((const float4*)p)[0];
  u.f[1] = ((const float4*)p)[1];
  return u.v;
}

// --- cooperative staging of a K x 64 B-tile: Wt -> sB[64][K+8] -------------
template <int K>
__device__ inline void fill_btile64(const __bf16* __restrict__ Wt, int colBase,
                                    int Hout, __bf16* sB, int tid) {
  constexpr int LDK = K + 8;      // 16B-granule pad: conflict-free column reads
  constexpr int chunks = K / 16;  // 32B chunks per column
  constexpr int total = 64 * chunks;
  for (int t = tid; t < total; t += 256) {
    const int c = t / chunks;
    const int k = (t % chunks) * 16;
    __bf16* d = sB + c * LDK + k;
    const int col = colBase + c;
    if (col < Hout) {
      const __bf16* g = Wt + (size_t)col * K + k;
#if HAVE_ASYNC_LDS
      async_cp16(g, d);
      async_cp16(g + 8, d + 8);
#else
      ((float4*)d)[0] = ((const float4*)g)[0];
      ((float4*)d)[1] = ((const float4*)g)[1];
#endif
    } else {
      const float4 z = {};
      ((float4*)d)[0] = z;
      ((float4*)d)[1] = z;
    }
  }
}

__device__ inline void btile_sync() {
#if HAVE_ASYNC_LDS
  wait_async0();
#endif
  __syncthreads();
}

// --- Generic GEMM: out = act(A[N,K] @ W + bias), W as bf16 Wt[Hout,K] ------
// grid: (ceil(3125/8), ceil(Hout/64)), block 256 = 8 wave32; 16x64 per wave.
template <int K, bool RELU>
__global__ void gemm_bias_kernel(const float* __restrict__ A,
                                 const __bf16* __restrict__ Wt,
                                 const float* __restrict__ bias,
                                 float* __restrict__ out, int Hout) {
  constexpr int LDK = K + 8;
  __shared__ __bf16 sB[64 * LDK];
  const int tid = threadIdx.x;
  const int colBase = blockIdx.y * 64;
  fill_btile64<K>(Wt, colBase, Hout, sB, tid);
  btile_sync();

  const int wave = tid >> 5, lane = tid & 31;
  const int mt = blockIdx.x * 8 + wave;
  if (mt >= kN / 16) return;          // wave-uniform; EXEC all-ones in WMMA
  const int row = mt * 16 + (lane & 15);
  const int half = lane >> 4;
  const __bf16* bbase = sB + (lane & 15) * LDK + 16 * half;
  v8f acc[4] = {{}, {}, {}, {}};
#pragma unroll
  for (int k0 = 0; k0 < K; k0 += 32) {
    v16bf a = load_a_frag(A, K, row, k0, lane, 1.0f);
#pragma unroll
    for (int s = 0; s < 4; ++s) {
      v16bf b = lds_b_frag(bbase + s * 16 * LDK + k0);
      acc[s] = __builtin_amdgcn_wmma_f32_16x16x32_bf16(false, a, false, b,
                                                       (short)0, acc[s],
                                                       false, false);
    }
  }
#pragma unroll
  for (int s = 0; s < 4; ++s) {
    const int col = colBase + 16 * s + (lane & 15);
    if (col >= Hout) continue;
    const float bv = bias[col];
#pragma unroll
    for (int r = 0; r < 8; ++r) {
      const int orow = mt * 16 + r + 8 * half;
      float v = acc[s][r] + bv;
      if (RELU) v = fmaxf(v, 0.0f);
      out[(size_t)orow * Hout + col] = v;
    }
  }
}

// --- Fused SAGE: out = relu(bn(mean@Wl + bl + x@Wr)) [+ res] ---------------
template <bool RES>
__global__ void sage_bn_kernel(const float* __restrict__ agg,
                               const float* __restrict__ cnt,
                               const float* __restrict__ xin,
                               const __bf16* __restrict__ WlT,
                               const __bf16* __restrict__ WrT,
                               const float* __restrict__ bl,
                               const float* __restrict__ bng,
                               const float* __restrict__ bnb,
                               const float* __restrict__ bnm,
                               const float* __restrict__ bnv,
                               const float* __restrict__ res,
                               float* __restrict__ out) {
  constexpr int K = kH, LDK = K + 8;
  __shared__ __bf16 sBl[64 * LDK];
  __shared__ __bf16 sBr[64 * LDK];
  const int tid = threadIdx.x;
  const int colBase = blockIdx.y * 64;
  fill_btile64<K>(WlT, colBase, kH, sBl, tid);
  fill_btile64<K>(WrT, colBase, kH, sBr, tid);
  btile_sync();

  const int wave = tid >> 5, lane = tid & 31;
  const int mt = blockIdx.x * 8 + wave;
  if (mt >= kN / 16) return;
  const int arow = mt * 16 + (lane & 15);
  const int half = lane >> 4;
  const float inv = 1.0f / fmaxf(cnt[arow], 1.0f);   // mean aggregation
  const __bf16* bpl = sBl + (lane & 15) * LDK + 16 * half;
  const __bf16* bpr = sBr + (lane & 15) * LDK + 16 * half;
  v8f acc[4] = {{}, {}, {}, {}};
#pragma unroll
  for (int k0 = 0; k0 < K; k0 += 32) {
    v16bf a1 = load_a_frag(agg, kH, arow, k0, lane, inv);
#pragma unroll
    for (int s = 0; s < 4; ++s) {
      v16bf b = lds_b_frag(bpl + s * 16 * LDK + k0);
      acc[s] = __builtin_amdgcn_wmma_f32_16x16x32_bf16(false, a1, false, b,
                                                       (short)0, acc[s],
                                                       false, false);
    }
    v16bf a2 = load_a_frag(xin, kH, arow, k0, lane, 1.0f);
#pragma unroll
    for (int s = 0; s < 4; ++s) {
      v16bf b = lds_b_frag(bpr + s * 16 * LDK + k0);
      acc[s] = __builtin_amdgcn_wmma_f32_16x16x32_bf16(false, a2, false, b,
                                                       (short)0, acc[s],
                                                       false, false);
    }
  }
#pragma unroll
  for (int s = 0; s < 4; ++s) {
    const int col = colBase + 16 * s + (lane & 15);
    const float sc = bng[col] * rsqrtf(bnv[col] + kEPS);
    const float sh = bnb[col] - bnm[col] * sc;
    const float bv = bl[col];
#pragma unroll
    for (int r = 0; r < 8; ++r) {
      const int orow = mt * 16 + r + 8 * half;
      float v = fmaxf((acc[s][r] + bv) * sc + sh, 0.0f);
      if (RES) v += res[(size_t)orow * kH + col];
      out[(size_t)orow * kH + col] = v;
    }
  }
}

// --- Gate: cur = sigmoid(prev@gW + gb) * prev + (1-sig) * cur --------------
__global__ void gate_kernel(const float* __restrict__ prev,
                            const __bf16* __restrict__ gWT,
                            const float* __restrict__ gb,
                            float* __restrict__ cur) {
  constexpr int K = kH, LDK = K + 8;
  __shared__ __bf16 sB[64 * LDK];
  const int tid = threadIdx.x;
  const int colBase = blockIdx.y * 64;
  fill_btile64<K>(gWT, colBase, kH, sB, tid);
  btile_sync();

  const int wave = tid >> 5, lane = tid & 31;
  const int mt = blockIdx.x * 8 + wave;
  if (mt >= kN / 16) return;
  const int row = mt * 16 + (lane & 15);
  const int half = lane >> 4;
  const __bf16* bbase = sB + (lane & 15) * LDK + 16 * half;
  v8f acc[4] = {{}, {}, {}, {}};
#pragma unroll
  for (int k0 = 0; k0 < K; k0 += 32) {
    v16bf a = load_a_frag(prev, kH, row, k0, lane, 1.0f);
#pragma unroll
    for (int s = 0; s < 4; ++s) {
      v16bf b = lds_b_frag(bbase + s * 16 * LDK + k0);
      acc[s] = __builtin_amdgcn_wmma_f32_16x16x32_bf16(false, a, false, b,
                                                       (short)0, acc[s],
                                                       false, false);
    }
  }
#pragma unroll
  for (int s = 0; s < 4; ++s) {
    const int col = colBase + 16 * s + (lane & 15);
    const float bv = gb[col];
#pragma unroll
    for (int r = 0; r < 8; ++r) {
      const int orow = mt * 16 + r + 8 * half;
      const size_t idx = (size_t)orow * kH + col;
      const float g = 1.0f / (1.0f + __expf(-(acc[s][r] + bv)));
      cur[idx] = g * prev[idx] + (1.0f - g) * cur[idx];
    }
  }
}

// --- Graph ops -------------------------------------------------------------
__global__ void degree_kernel(const long long* __restrict__ dst,
                              float* __restrict__ cnt) {
  const int e = blockIdx.x * blockDim.x + threadIdx.x;
  if (e < kE) atomicAdd(&cnt[(int)dst[e]], 1.0f);
}

// agg[dst] += x[src]; feature matrix (51MB) is L2-resident (192MB L2).
__global__ void scatter_kernel(const float* __restrict__ x,
                               const long long* __restrict__ src,
                               const long long* __restrict__ dst,
                               float* __restrict__ agg) {
  const long long tid = (long long)blockIdx.x * blockDim.x + threadIdx.x;
  if (tid >= (long long)kE * (kH / 4)) return;
  const int e  = (int)(tid >> 6);          // kH/4 == 64 chunks per edge
  const int f4 = ((int)tid & 63) * 4;
  const int s = (int)src[e], d = (int)dst[e];
  const float4 v = *(const float4*)(x + (size_t)s * kH + f4);
  float* ap = agg + (size_t)d * kH + f4;
  atomicAdd(ap + 0, v.x); atomicAdd(ap + 1, v.y);
  atomicAdd(ap + 2, v.z); atomicAdd(ap + 3, v.w);
}

// ---------------------------------------------------------------------------
extern "C" void kernel_launch(void* const* d_in, const int* in_sizes, int n_in,
                              void* d_out, int out_size, void* d_ws, size_t ws_size,
                              hipStream_t stream) {
  const float*     x   = (const float*)d_in[0];
  const long long* ei  = (const long long*)d_in[1];   // int64 per reference
  const long long* src = ei;
  const long long* dst = ei + kE;
  const float* Wp  = (const float*)d_in[2];
  const float* bp  = (const float*)d_in[3];
  const float* W1l = (const float*)d_in[4];
  const float* b1l = (const float*)d_in[5];
  const float* W1r = (const float*)d_in[6];
  const float* W2l = (const float*)d_in[7];
  const float* b2l = (const float*)d_in[8];
  const float* W2r = (const float*)d_in[9];
  const float* bn1g = (const float*)d_in[10];
  const float* bn1b = (const float*)d_in[11];
  const float* bn1m = (const float*)d_in[12];
  const float* bn1v = (const float*)d_in[13];
  const float* bn2g = (const float*)d_in[14];
  const float* bn2b = (const float*)d_in[15];
  const float* bn2m = (const float*)d_in[16];
  const float* bn2v = (const float*)d_in[17];
  const float* gW  = (const float*)d_in[18];
  const float* gb  = (const float*)d_in[19];
  const float* Wc1 = (const float*)d_in[20];
  const float* bc1 = (const float*)d_in[21];
  const float* Wc2 = (const float*)d_in[22];
  const float* bc2 = (const float*)d_in[23];
  float* out = (float*)d_out;

  const size_t NH = (size_t)kN * kH;
  float* ws   = (float*)d_ws;
  float* buf0 = ws;             // prev
  float* buf1 = ws + NH;        // h1 (later classifier hidden)
  float* buf2 = ws + 2 * NH;    // cur
  float* agg  = ws + 3 * NH;
  float* cnt  = ws + 4 * NH;

  // bf16 transposed weights live after the f32 region (16B aligned).
  __bf16* bw = (__bf16*)(ws + 4 * NH + kN);
  const size_t HH = (size_t)kH * kH;
  __bf16* WpT  = bw;                       // [256,128]
  __bf16* W1lT = WpT  + (size_t)kH * kD;   // 3 x [256,256]
  __bf16* W1rT = W1lT + 3 * HH;
  __bf16* W2lT = W1rT + 3 * HH;
  __bf16* W2rT = W2lT + 3 * HH;
  __bf16* gWT  = W2rT + 3 * HH;            // 2 x [256,256]
  __bf16* Wc1T = gWT  + 2 * HH;            // [128,256]
  __bf16* Wc2T = Wc1T + (size_t)kD * kH;   // [40,128]

  const int TPB = 256;
  auto tb = [](int n) { return (n + 255) / 256; };

  // ---- weight prep (tiny; ~2MB of bf16) ----
  transpose_bf16_kernel<<<tb(kD * kH), TPB, 0, stream>>>(Wp, WpT, kD, kH);
  for (int i = 0; i < kB; ++i) {
    transpose_bf16_kernel<<<tb(kH * kH), TPB, 0, stream>>>(W1l + i * HH, W1lT + i * HH, kH, kH);
    transpose_bf16_kernel<<<tb(kH * kH), TPB, 0, stream>>>(W1r + i * HH, W1rT + i * HH, kH, kH);
    transpose_bf16_kernel<<<tb(kH * kH), TPB, 0, stream>>>(W2l + i * HH, W2lT + i * HH, kH, kH);
    transpose_bf16_kernel<<<tb(kH * kH), TPB, 0, stream>>>(W2r + i * HH, W2rT + i * HH, kH, kH);
  }
  for (int i = 0; i < 2; ++i)
    transpose_bf16_kernel<<<tb(kH * kH), TPB, 0, stream>>>(gW + i * HH, gWT + i * HH, kH, kH);
  transpose_bf16_kernel<<<tb(kH * kD), TPB, 0, stream>>>(Wc1, Wc1T, kH, kD);
  transpose_bf16_kernel<<<tb(kD * kC), TPB, 0, stream>>>(Wc2, Wc2T, kD, kC);

  // ---- degree (graph static across layers) ----
  (void)hipMemsetAsync(cnt, 0, kN * sizeof(float), stream);
  degree_kernel<<<tb(kE), TPB, 0, stream>>>(dst, cnt);

  const unsigned mblk = (kN / 16 + 7) / 8;        // 391
  const dim3 gridH(mblk, kH / 64);                // Hout=256 -> y=4
  const dim3 gridD(mblk, kD / 64);                // Hout=128 -> y=2
  const dim3 gridC(mblk, 1);                      // Hout=40  -> y=1 (masked)

  // ---- input projection: prev = x @ Wp + bp ----
  gemm_bias_kernel<kD, false><<<gridH, TPB, 0, stream>>>(x, WpT, bp, buf0, kH);

  float* prev  = buf0;
  float* other = buf2;
  const long long scatThreads = (long long)kE * (kH / 4);
  const unsigned scatBlocks = (unsigned)((scatThreads + TPB - 1) / TPB);

  for (int i = 0; i < kB; ++i) {
    // h1 = relu(bn1(sage(prev)))
    (void)hipMemsetAsync(agg, 0, NH * sizeof(float), stream);
    scatter_kernel<<<scatBlocks, TPB, 0, stream>>>(prev, src, dst, agg);
    sage_bn_kernel<false><<<gridH, TPB, 0, stream>>>(
        agg, cnt, prev, W1lT + i * HH, W1rT + i * HH, b1l + i * kH,
        bn1g + i * kH, bn1b + i * kH, bn1m + i * kH, bn1v + i * kH,
        nullptr, buf1);

    // cur = relu(bn2(sage(h1))) + prev
    (void)hipMemsetAsync(agg, 0, NH * sizeof(float), stream);
    scatter_kernel<<<scatBlocks, TPB, 0, stream>>>(buf1, src, dst, agg);
    sage_bn_kernel<true><<<gridH, TPB, 0, stream>>>(
        agg, cnt, buf1, W2lT + i * HH, W2rT + i * HH, b2l + i * kH,
        bn2g + i * kH, bn2b + i * kH, bn2m + i * kH, bn2v + i * kH,
        prev, other);

    // highway gate for blocks 1..B-1
    if (i > 0)
      gate_kernel<<<gridH, TPB, 0, stream>>>(prev, gWT + (size_t)(i - 1) * HH,
                                             gb + (size_t)(i - 1) * kH, other);
    float* t = prev; prev = other; other = t;
  }

  // ---- classifier: out = relu(prev@Wc1 + bc1) @ Wc2 + bc2 ----
  float* hid = buf1;  // [N,128]
  gemm_bias_kernel<kH, true ><<<gridD, TPB, 0, stream>>>(prev, Wc1T, bc1, hid, kD);
  gemm_bias_kernel<kD, false><<<gridC, TPB, 0, stream>>>(hid, Wc2T, bc2, out, kC);
}
